// GraphAttention_13022340841592
// MI455X (gfx1250) — matrix operand order repaired and checked by reading, not compiled, not run on hardware
//
#include <hip/hip_runtime.h>

typedef __attribute__((ext_vector_type(16))) _Float16 v16h;
typedef __attribute__((ext_vector_type(8)))  _Float16 v8h;
typedef __attribute__((ext_vector_type(8)))  float    v8f;

#define B_   4
#define N_   4096
#define FIN_ 128
#define H_   4
#define U_   32
#define HU_  128
#define ST_  4104   // padded LDS row stride in halves (8208 B -> bank-conflict-free A tiles)

// ---------------------------------------------------------------------------
// Kernel 1: h = x@W (f32), store h transposed [B,H,U,N] as f16 for WMMA B
// operand, and compute f1/f2 = <h, a_src/a_dst> per (b, head, n) via wave32
// reductions (one wave == one head since U == warpSize == 32).
// ---------------------------------------------------------------------------
__global__ __launch_bounds__(128) void gat_proj(
    const float* __restrict__ x, const float* __restrict__ W,
    const float* __restrict__ a_src, const float* __restrict__ a_dst,
    float* __restrict__ f1, float* __restrict__ f2,
    _Float16* __restrict__ ht)
{
    __shared__ float xs[FIN_];
    const int r = blockIdx.x;            // b*N + n
    const int b = r >> 12;               // N = 4096
    const int n = r & (N_ - 1);
    const int f = threadIdx.x;           // 0..127 feature
    xs[f] = x[(size_t)r * FIN_ + f];
    __syncthreads();

    float acc = 0.f;
    #pragma unroll 8
    for (int k = 0; k < FIN_; ++k)
        acc = fmaf(xs[k], W[k * HU_ + f], acc);

    // transposed f16 store: row (b*128+f) == (b, head, u), column n
    ht[(size_t)(b * HU_ + f) * N_ + n] = (_Float16)acc;

    const int head = f >> 5;
    float s1 = acc * a_src[f];
    float s2 = acc * a_dst[f];
    #pragma unroll
    for (int off = 16; off; off >>= 1) {
        s1 += __shfl_xor(s1, off, 32);
        s2 += __shfl_xor(s2, off, 32);
    }
    if ((f & 31) == 0) {
        f1[(size_t)(b * H_ + head) * N_ + n] = s1;
        f2[(size_t)(b * H_ + head) * N_ + n] = s2;
    }
}

// ---------------------------------------------------------------------------
// Kernel 2: one 8-wave block per (b, head, 16-row tile). Scores for the
// 16x4096 tile are built in LDS (f16), softmaxed in two passes, then
// alpha @ h is computed with v_wmma_f32_16x16x32_f16 (A from LDS, B from
// global f16 h-transpose which is L2 resident). Partials reduced with
// ds_add_f32; epilogue applies 1/rowsum, bias, ELU.
// ---------------------------------------------------------------------------
__global__ __launch_bounds__(256) void gat_attn(
    const float* __restrict__ adj, const float* __restrict__ f1,
    const float* __restrict__ f2, const _Float16* __restrict__ ht,
    const float* __restrict__ bias, float* __restrict__ out)
{
    extern __shared__ char smem[];
    _Float16* et = (_Float16*)smem;                    // 16 rows x ST_ halves
    float* rmax  = (float*)(smem + 16 * ST_ * 2);      // 16
    float* rsum  = rmax + 16;                          // 16
    float* red   = rsum + 16;                          // 2 tiles x 256 f32

    const int it   = blockIdx.x & 255;                 // N/16 = 256 tiles
    const int head = (blockIdx.x >> 8) & (H_ - 1);
    const int b    = blockIdx.x >> 10;
    const int bh   = b * H_ + head;
    const int i0   = it * 16;

    const int tid  = threadIdx.x;
    const int wave = tid >> 5;
    const int lane = tid & 31;

    // zero cross-wave reduction buffer
    red[tid] = 0.f;
    red[tid + 256] = 0.f;

    // ---- phase 1: e = mask(LeakyReLU(f1_i + f2_j)); store f16; row max ----
    const float* f2row = f2 + (size_t)bh * N_;
    for (int rr = wave * 2; rr < wave * 2 + 2; ++rr) {
        const int i = i0 + rr;
        const float f1v = f1[(size_t)bh * N_ + i];
        const float* arow = adj + ((size_t)b * N_ + i) * N_;
        float m = -3.4e38f;
        for (int j = lane; j < N_; j += 32) {
            float e = f1v + f2row[j];
            e = e > 0.f ? e : 0.2f * e;                // LeakyReLU(0.2)
            const bool on = (arow[j] > 0.f) || (j == i);  // adj + self-loop
            e = on ? e : -1e9f;
            m = fmaxf(m, e);
            et[rr * ST_ + j] = (_Float16)e;
        }
        #pragma unroll
        for (int off = 16; off; off >>= 1) m = fmaxf(m, __shfl_xor(m, off, 32));
        if (lane == 0) rmax[rr] = m;
    }
    __syncthreads();

    // ---- phase 2: alpha_unnorm = exp(e - max); row sum ----
    for (int rr = wave * 2; rr < wave * 2 + 2; ++rr) {
        const float m = rmax[rr];
        float s = 0.f;
        for (int j = lane; j < N_; j += 32) {
            float p = __expf((float)et[rr * ST_ + j] - m);
            et[rr * ST_ + j] = (_Float16)p;
            s += p;
        }
        #pragma unroll
        for (int off = 16; off; off >>= 1) s += __shfl_xor(s, off, 32);
        if (lane == 0) rsum[rr] = s;
    }
    __syncthreads();

    // ---- phase 3: C += A(16x32 f16, LDS) x B(32x16 f16, global) ----
    const int M  = lane & 15;   // A row / B column / C column
    const int hi = lane >> 4;   // lane half selects K half (k = 16*hi + s)
    const _Float16* hb = ht + (size_t)bh * U_ * N_;
    v8f c0 = {};
    v8f c1 = {};
    const int jw = wave * 512;  // this wave's K slice
    for (int s = 0; s < 16; ++s) {
        const int j0 = jw + s * 32;
        // A, contiguous K order: lanes 0-15 K=0-15, lanes 16-31 K=16-31 (matches B)
        v8h alo = *(const v8h*)(et + M * ST_ + j0 + hi * 16);
        v8h ahi = *(const v8h*)(et + M * ST_ + j0 + hi * 16 + 8);
        v16h a;
        #pragma unroll
        for (int t = 0; t < 8; ++t) { a[t] = alo[t]; a[8 + t] = ahi[t]; }
        // B, same contiguous K order: lanes 0-15 K=0-15, lanes 16-31 K=16-31; N = lane%16
        const _Float16* b0p = hb + (size_t)M * N_ + j0 + hi * 16;
        const _Float16* b1p = hb + (size_t)(16 + M) * N_ + j0 + hi * 16;
        v8h bl0 = *(const v8h*)b0p;
        v8h bh0 = *(const v8h*)(b0p + 8);
        v8h bl1 = *(const v8h*)b1p;
        v8h bh1 = *(const v8h*)(b1p + 8);
        v16h b0, b1;
        #pragma unroll
        for (int t = 0; t < 8; ++t) {
            b0[t] = bl0[t]; b0[8 + t] = bh0[t];
            b1[t] = bl1[t]; b1[8 + t] = bh1[t];
        }
        c0 = __builtin_amdgcn_wmma_f32_16x16x32_f16(false, a, false, b0,
                                                    (short)0, c0, false, false);
        c1 = __builtin_amdgcn_wmma_f32_16x16x32_f16(false, a, false, b1,
                                                    (short)0, c1, false, false);
    }

    // cross-wave reduction: C element (vgpr c, lane) = row c+8*hi, col lane%16
    #pragma unroll
    for (int c = 0; c < 8; ++c) {
        atomicAdd(&red[(c + 8 * hi) * 16 + M], c0[c]);
        atomicAdd(&red[256 + (c + 8 * hi) * 16 + M], c1[c]);
    }
    __syncthreads();

    // ---- epilogue: 1/rowsum, +bias, ELU, store [B,N,H*U] ----
    for (int idx = tid; idx < 512; idx += 256) {
        const int row = idx >> 5;
        const int u   = idx & 31;
        float v = red[(u >> 4) * 256 + row * 16 + (u & 15)] / rsum[row];
        v += bias[head * U_ + u];
        v = v > 0.f ? v : expm1f(v);   // ELU
        out[((size_t)(b * N_ + i0 + row)) * HU_ + head * U_ + u] = v;
    }
}

// ---------------------------------------------------------------------------
extern "C" void kernel_launch(void* const* d_in, const int* in_sizes, int n_in,
                              void* d_out, int out_size, void* d_ws, size_t ws_size,
                              hipStream_t stream) {
    (void)in_sizes; (void)n_in; (void)out_size; (void)ws_size;
    const float* x     = (const float*)d_in[0];
    const float* adj   = (const float*)d_in[1];
    const float* W     = (const float*)d_in[2];
    const float* a_src = (const float*)d_in[3];
    const float* a_dst = (const float*)d_in[4];
    const float* bias  = (const float*)d_in[5];
    float* out = (float*)d_out;

    // workspace: f1 [B*H*N] f32 | f2 [B*H*N] f32 | ht [B*H*U*N] f16
    float* f1 = (float*)d_ws;
    float* f2 = f1 + (size_t)B_ * H_ * N_;
    _Float16* ht = (_Float16*)(f2 + (size_t)B_ * H_ * N_);

    const size_t shmem2 = (size_t)16 * ST_ * 2 + 32 * sizeof(float) + 512 * sizeof(float);
    hipFuncSetAttribute((const void*)gat_attn,
                        hipFuncAttributeMaxDynamicSharedMemorySize, (int)shmem2);

    gat_proj<<<B_ * N_, 128, 0, stream>>>(x, W, a_src, a_dst, f1, f2, ht);
    gat_attn<<<B_ * H_ * (N_ / 16), 256, shmem2, stream>>>(adj, f1, f2, ht, bias, out);
}
